// GCNWithEdgeFeatures_39951785787697
// MI455X (gfx1250) — matrix operand order repaired and checked by reading, not compiled, not run on hardware
//
#include <hip/hip_runtime.h>
#include <stdint.h>

#define E_EDGES 2048
#define N_NODES 1024
#define N_GRAPHS 64
#define WG 256
#define MT 256      // edge-tile (M)
#define KCH 256     // K staged in LDS per chunk
#define BPAD (KCH + 8)
#define KSPLIT 2048 // K per blockIdx.z slice

typedef __attribute__((ext_vector_type(16))) __bf16 v16bf;
typedef __attribute__((ext_vector_type(8)))  float  v8f;

union Frag16 { v16bf v; uint32_t u32[8]; uint16_t u16[16]; };

__device__ __forceinline__ uint16_t f32_bf16(float f) {
  uint32_t u = __float_as_uint(f);
  u += 0x7fffu + ((u >> 16) & 1u);   // round-to-nearest-even
  return (uint16_t)(u >> 16);
}

// h = relu(edge_attr @ w1 + b1), stored bf16, EF=4 hardcoded
__global__ void k_edge_hidden(const float* __restrict__ EA, const float* __restrict__ W1,
                              const float* __restrict__ B1, uint16_t* __restrict__ H,
                              int K, int total) {
  int idx = blockIdx.x * blockDim.x + threadIdx.x;
  if (idx >= total) return;
  int e = idx / K, j = idx - e * K;
  const float* ea = EA + e * 4;
  float s = B1[j];
#pragma unroll
  for (int f = 0; f < 4; ++f) s += ea[f] * W1[f * K + j];
  H[idx] = f32_bf16(fmaxf(s, 0.f));
}

// msg[e,o] = sum_i x[src[e],i] * b2[i*dout+o]   (bias part of fused contraction)
__global__ void k_msg_bias(const float* __restrict__ X, const int* __restrict__ src,
                           const float* __restrict__ B2, float* __restrict__ msg,
                           int din, int dout, int total) {
  int idx = blockIdx.x * blockDim.x + threadIdx.x;
  if (idx >= total) return;
  int e = idx / dout, o = idx - e * dout;
  const float* xr = X + src[e] * din;
  float s = 0.f;
  for (int i = 0; i < din; ++i) s += xr[i] * B2[i * dout + o];
  msg[idx] = s;
}

// msg[e,o] += sum_i x[src[e],i] * (H[e,:] @ W2[:, i*DOUT+o])   via bf16 WMMA
template <int K, int DIN, int DOUT>
__global__ __launch_bounds__(WG)
void k_fused_msg(const uint16_t* __restrict__ H, const float* __restrict__ W2,
                 const float* __restrict__ X, const int* __restrict__ src,
                 float* __restrict__ msg) {
  constexpr size_t NCOLS = (size_t)DIN * DOUT;
  extern __shared__ char smem[];
  uint16_t* As = (uint16_t*)smem;                            // [MT][KCH] bf16
  uint16_t* Bs = (uint16_t*)(smem + MT * KCH * 2);           // [16][BPAD] bf16 (col-major)
  float*    Xs = (float*)(smem + MT * KCH * 2 + 16 * BPAD * 2); // [MT][DIN]

  const int tid  = threadIdx.x;
  const int wave = tid >> 5;
  const int lane = tid & 31;
  const int ln   = lane & 15;
  const int hi   = lane >> 4;            // 0 | 1 (lane half)
  const int e0   = blockIdx.x * MT;
  const int o0   = blockIdx.y * 16;

  for (int idx = tid; idx < MT * DIN; idx += WG) {
    int r = idx / DIN, i = idx - r * DIN;
    Xs[idx] = X[src[e0 + r] * DIN + i];
  }

  v8f acc0 = {}, acc1 = {};
  const int row0 = wave * 32 + ln;       // wave owns row-tiles 2w, 2w+1
  const int row1 = row0 + 16;

  const int kbeg = blockIdx.z * KSPLIT;
  const int kend = (K < kbeg + KSPLIT) ? K : (kbeg + KSPLIT);

  for (int k0 = kbeg; k0 < kend; k0 += KCH) {
    const int krem = kend - k0;
    const int kbs  = ((krem < KCH) ? krem : KCH) / 32;  // 32-K blocks staged
    __syncthreads();
    const int n8 = kbs * 4;                              // 16B chunks / row of A
    // ---- async copy A tile (bf16 raw bytes) global -> LDS, tracked by ASYNCcnt
    for (int idx = tid; idx < MT * n8; idx += WG) {
      int r = idx / n8, c = idx - r * n8;
      const uint16_t* gptr = H + (size_t)(e0 + r) * K + k0 + c * 8;
      uint32_t ldsoff = (uint32_t)(uintptr_t)(As + r * KCH + c * 8);
      asm volatile("global_load_async_to_lds_b128 %0, %1, off"
                   :: "v"(ldsoff), "v"(gptr) : "memory");
    }
    asm volatile("s_wait_asynccnt 0" ::: "memory");
    __syncthreads();

    for (int i = 0; i < DIN; ++i) {
      // ---- stage B block [kbs*32 x 16] f32 -> bf16 LDS (shared by all 8 waves)
      __syncthreads();
      const float* gb = W2 + (size_t)k0 * NCOLS + i * DOUT + o0;
      const int nel = kbs * 32 * 16;
      for (int idx = tid; idx < nel; idx += WG) {
        const int col = idx & 15, kk = idx >> 4;
        Bs[col * BPAD + kk] = f32_bf16(gb[(size_t)kk * NCOLS + col]);
      }
      if (i + 1 < DIN) {  // prefetch next-i block: first 16 K rows, one 64B line per 16 threads
        __builtin_prefetch(gb + DOUT + (size_t)(tid >> 4) * NCOLS + (tid & 15), 0, 1);
      }
      __syncthreads();

      v8f c0 = {}, c1 = {};
      for (int kb = 0; kb < kbs; ++kb) {
        Frag16 a0, a1, b;
        const int ka = kb * 32 + hi * 8;   // A: K {0..7,16..23} | {8..15,24..31}
        *(uint4*)(&a0.u32[0]) = *(const uint4*)(As + row0 * KCH + ka);
        *(uint4*)(&a0.u32[4]) = *(const uint4*)(As + row0 * KCH + ka + 16);
        *(uint4*)(&a1.u32[0]) = *(const uint4*)(As + row1 * KCH + ka);
        *(uint4*)(&a1.u32[4]) = *(const uint4*)(As + row1 * KCH + ka + 16);
        // B: lanes 0-15 -> K {0..7,16..23}, lanes 16-31 -> K {8..15,24..31}, col = lane&15
        const int kbB = kb * 32 + hi * 8;
        *(uint4*)(&b.u32[0]) = *(const uint4*)(Bs + ln * BPAD + kbB);
        *(uint4*)(&b.u32[4]) = *(const uint4*)(Bs + ln * BPAD + kbB + 16);
        c0 = __builtin_amdgcn_wmma_f32_16x16x32_bf16(false, a0.v, false, b.v,
                                                     (short)0, c0, false, false);
        c1 = __builtin_amdgcn_wmma_f32_16x16x32_bf16(false, a1.v, false, b.v,
                                                     (short)0, c1, false, false);
      }
#pragma unroll
      for (int r = 0; r < 8; ++r) {   // acc += x[src[e],i] * C  (row-scale)
        acc0[r] += Xs[(wave * 32 + r + hi * 8) * DIN + i] * c0[r];
        acc1[r] += Xs[(wave * 32 + 16 + r + hi * 8) * DIN + i] * c1[r];
      }
    }
  }

#pragma unroll
  for (int r = 0; r < 8; ++r) {
    atomicAdd(&msg[(size_t)(e0 + wave * 32 + r + hi * 8) * DOUT + o0 + ln], acc0[r]);
    atomicAdd(&msg[(size_t)(e0 + wave * 32 + 16 + r + hi * 8) * DOUT + o0 + ln], acc1[r]);
  }
}

template <int K, int DIN, int DOUT>
static void launch_fused(const uint16_t* h, const float* w2, const float* xin,
                         const int* src, float* msg, hipStream_t stream) {
  size_t smem = (size_t)MT * KCH * 2 + 16 * BPAD * 2 + (size_t)MT * DIN * 4;
  dim3 grid(E_EDGES / MT, DOUT / 16, (K + KSPLIT - 1) / KSPLIT);
  k_fused_msg<K, DIN, DOUT><<<grid, WG, smem, stream>>>(h, w2, xin, src, msg);
}

__global__ void k_count(const int* __restrict__ dst, float* __restrict__ cnt, int E) {
  int e = blockIdx.x * blockDim.x + threadIdx.x;
  if (e < E) atomicAdd(&cnt[dst[e]], 1.f);
}

__global__ void k_scatter(const float* __restrict__ msg, const int* __restrict__ dst,
                          float* __restrict__ s, int dout, int total) {
  int idx = blockIdx.x * blockDim.x + threadIdx.x;
  if (idx >= total) return;
  int e = idx / dout, o = idx - e * dout;
  atomicAdd(&s[dst[e] * dout + o], msg[idx]);
}

// xout = relu(s/max(cnt,1) + xin @ root + bias)
__global__ void k_combine(const float* __restrict__ s, const float* __restrict__ cnt,
                          const float* __restrict__ Xin, const float* __restrict__ root,
                          const float* __restrict__ bias, float* __restrict__ Xout,
                          int din, int dout, int total) {
  int idx = blockIdx.x * blockDim.x + threadIdx.x;
  if (idx >= total) return;
  int n = idx / dout, o = idx - n * dout;
  float v = s[idx] / fmaxf(cnt[n], 1.f);
  const float* xr = Xin + n * din;
  for (int k = 0; k < din; ++k) v += xr[k] * root[k * dout + o];
  v += bias[o];
  Xout[idx] = fmaxf(v, 0.f);
}

__global__ void k_pool_cnt(const int* __restrict__ batch, float* __restrict__ gc, int N) {
  int n = blockIdx.x * blockDim.x + threadIdx.x;
  if (n < N) atomicAdd(&gc[batch[n]], 1.f);
}
__global__ void k_pool_add(const float* __restrict__ x, const int* __restrict__ batch,
                           float* __restrict__ gs, int dout, int total) {
  int idx = blockIdx.x * blockDim.x + threadIdx.x;
  if (idx >= total) return;
  int n = idx / dout, o = idx - n * dout;
  atomicAdd(&gs[batch[n] * dout + o], x[idx]);
}
__global__ void k_pool_div(float* __restrict__ gs, const float* __restrict__ gc,
                           int dout, int total) {
  int idx = blockIdx.x * blockDim.x + threadIdx.x;
  if (idx >= total) return;
  gs[idx] = gs[idx] / fmaxf(gc[idx / dout], 1.f);
}

// out[m,n] = act(A[m,:] @ W[:,n] + b[n])
__global__ void k_linear(const float* __restrict__ A, const float* __restrict__ W,
                         const float* __restrict__ b, float* __restrict__ out,
                         int Kd, int Nd, int relu, int total) {
  int idx = blockIdx.x * blockDim.x + threadIdx.x;
  if (idx >= total) return;
  int m = idx / Nd, n = idx - m * Nd;
  float s = b[n];
  const float* a = A + m * Kd;
  for (int k = 0; k < Kd; ++k) s += a[k] * W[k * Nd + n];
  out[idx] = relu ? fmaxf(s, 0.f) : s;
}

static inline int cdiv(int a, int b) { return (a + b - 1) / b; }

extern "C" void kernel_launch(void* const* d_in, const int* in_sizes, int n_in,
                              void* d_out, int out_size, void* d_ws, size_t ws_size,
                              hipStream_t stream) {
  const float* x   = (const float*)d_in[0];
  const int*   ei  = (const int*)d_in[1];
  const float* ea  = (const float*)d_in[2];
  const int*   bat = (const int*)d_in[3];
  const float* en_w1[3] = {(const float*)d_in[4],  (const float*)d_in[8],  (const float*)d_in[12]};
  const float* en_b1[3] = {(const float*)d_in[5],  (const float*)d_in[9],  (const float*)d_in[13]};
  const float* en_w2[3] = {(const float*)d_in[6],  (const float*)d_in[10], (const float*)d_in[14]};
  const float* en_b2[3] = {(const float*)d_in[7],  (const float*)d_in[11], (const float*)d_in[15]};
  const float* roots[3] = {(const float*)d_in[16], (const float*)d_in[18], (const float*)d_in[20]};
  const float* biases[3]= {(const float*)d_in[17], (const float*)d_in[19], (const float*)d_in[21]};
  const float* fcw[4] = {(const float*)d_in[22], (const float*)d_in[24], (const float*)d_in[26], (const float*)d_in[28]};
  const float* fcb[4] = {(const float*)d_in[23], (const float*)d_in[25], (const float*)d_in[27], (const float*)d_in[29]};
  const int* src = ei;
  const int* dst = ei + E_EDGES;

  uint8_t* p = (uint8_t*)d_ws;
  auto alloc = [&](size_t bytes) { void* q = (void*)p; p += (bytes + 255) & ~(size_t)255; return q; };
  uint16_t* h = (uint16_t*)alloc((size_t)E_EDGES * 8192 * 2);  // bf16 edge hidden (max K)
  float* msg  = (float*)alloc((size_t)E_EDGES * 256 * 4);
  float* sbuf = (float*)alloc((size_t)N_NODES * 256 * 4);
  float* cnt  = (float*)alloc((size_t)N_NODES * 4);
  float* x1   = (float*)alloc((size_t)N_NODES * 64 * 4);
  float* x2   = (float*)alloc((size_t)N_NODES * 128 * 4);
  float* x3   = (float*)alloc((size_t)N_NODES * 256 * 4);
  float* gbuf = (float*)alloc((size_t)N_GRAPHS * 256 * 4);
  float* gcnt = (float*)alloc((size_t)N_GRAPHS * 4);
  float* f1   = (float*)alloc((size_t)N_GRAPHS * 1024 * 4);
  float* f2   = (float*)alloc((size_t)N_GRAPHS * 1024 * 4);
  float* f3   = (float*)alloc((size_t)N_GRAPHS * 512 * 4);

  hipMemsetAsync(cnt, 0, (size_t)N_NODES * 4, stream);
  k_count<<<cdiv(E_EDGES, 256), 256, 0, stream>>>(dst, cnt, E_EDGES);

  const int Ks[3]    = {576, 4096, 8192};
  const int dins[3]  = {9, 64, 128};
  const int douts[3] = {64, 128, 256};
  const float* xin_l[3] = {x, x1, x2};
  float* xout_l[3]      = {x1, x2, x3};

  for (int L = 0; L < 3; ++L) {
    const int K = Ks[L], din = dins[L], dout = douts[L];
    int tot;
    tot = E_EDGES * K;
    k_edge_hidden<<<cdiv(tot, 256), 256, 0, stream>>>(ea, en_w1[L], en_b1[L], h, K, tot);
    tot = E_EDGES * dout;
    k_msg_bias<<<cdiv(tot, 256), 256, 0, stream>>>(xin_l[L], src, en_b2[L], msg, din, dout, tot);
    if (L == 0)      launch_fused< 576,   9,  64>(h, en_w2[L], xin_l[L], src, msg, stream);
    else if (L == 1) launch_fused<4096,  64, 128>(h, en_w2[L], xin_l[L], src, msg, stream);
    else             launch_fused<8192, 128, 256>(h, en_w2[L], xin_l[L], src, msg, stream);
    hipMemsetAsync(sbuf, 0, (size_t)N_NODES * dout * 4, stream);
    tot = E_EDGES * dout;
    k_scatter<<<cdiv(tot, 256), 256, 0, stream>>>(msg, dst, sbuf, dout, tot);
    tot = N_NODES * dout;
    k_combine<<<cdiv(tot, 256), 256, 0, stream>>>(sbuf, cnt, xin_l[L], roots[L], biases[L],
                                                  xout_l[L], din, dout, tot);
  }

  // global mean pool
  hipMemsetAsync(gbuf, 0, (size_t)N_GRAPHS * 256 * 4, stream);
  hipMemsetAsync(gcnt, 0, (size_t)N_GRAPHS * 4, stream);
  k_pool_cnt<<<cdiv(N_NODES, 256), 256, 0, stream>>>(bat, gcnt, N_NODES);
  int tot = N_NODES * 256;
  k_pool_add<<<cdiv(tot, 256), 256, 0, stream>>>(x3, bat, gbuf, 256, tot);
  tot = N_GRAPHS * 256;
  k_pool_div<<<cdiv(tot, 256), 256, 0, stream>>>(gbuf, gcnt, 256, tot);

  // FC head (small: scalar f32)
  tot = N_GRAPHS * 1024;
  k_linear<<<cdiv(tot, 256), 256, 0, stream>>>(gbuf, fcw[0], fcb[0], f1, 256, 1024, 1, tot);
  k_linear<<<cdiv(tot, 256), 256, 0, stream>>>(f1, fcw[1], fcb[1], f2, 1024, 1024, 1, tot);
  tot = N_GRAPHS * 512;
  k_linear<<<cdiv(tot, 256), 256, 0, stream>>>(f2, fcw[2], fcb[2], f3, 1024, 512, 1, tot);
  tot = N_GRAPHS * 3;
  k_linear<<<cdiv(tot, 256), 256, 0, stream>>>(f3, fcw[3], fcb[3], (float*)d_out, 512, 3, 0, tot);
}
